// RangeAttentionBlock_34282428956879
// MI455X (gfx1250) — matrix operand order repaired and checked by reading, not compiled, not run on hardware
//
#include <hip/hip_runtime.h>

typedef __attribute__((ext_vector_type(16))) _Float16 v16h;
typedef __attribute__((ext_vector_type(8)))  _Float16 h8;
typedef __attribute__((ext_vector_type(8)))  float    v8f;

namespace {
constexpr int kC     = 64;
constexpr int kCav   = 5;
constexpr int kHeads = 4;
constexpr int kHW    = 96 * 192;        // 18432 pixels
constexpr int kPT    = 16;              // pixels per workgroup tile
constexpr int kCols  = kCav * kPT;      // 80 columns (agent-major blocks of 16 pixels)
constexpr int kXS    = 72;              // bn(x) f16 column stride (halves, 16B-mult)
constexpr int kQS    = 194;             // QKV f16 column stride (halves)
constexpr int kOS    = 72;              // attn-out / mlp-in f16 column stride (halves)
constexpr int kPS    = 66;              // proj f32 column stride (floats)
constexpr float kEps = 1e-5f;

// Pre-swizzled weight fragments in workspace: tile = (mt*KT + kt), each tile is
// 32 lanes x 16 halves in exact CDNA5 A-fragment order (2x b128 per lane).
constexpr int kQT = 96;                 // Wqkv: 48 mt x 2 kt
constexpr int kOT = 32;                 // Wo:    4 mt x 8 kt
constexpr int kMT = 8;                  // Wmlp:  4 mt x 2 kt
constexpr int kWqkvOff = 0;
constexpr int kWoOff   = kQT * 512;
constexpr int kWmlpOff = kWoOff + kOT * 512;
constexpr int kWsBytes = (kQT + kOT + kMT) * 512 * 2;   // 139264 B

__device__ inline v8f wmma16x16x32(v16h a, v16h b, v8f c) {
  return __builtin_amdgcn_wmma_f32_16x16x32_f16(false, a, false, b, (short)0, c,
                                                false, false);
}

// A fragment from pre-swizzled f16 workspace (32B aligned -> 2x global b128).
__device__ inline v16h load_a_ws(const _Float16* __restrict__ Wf, int KT, int mt,
                                 int kt, int lane) {
  const _Float16* p = Wf + (((mt * KT + kt) * 32) + lane) * 16;
  return *(const v16h*)p;
}

// Fallback A fragment from row-major f32 weights (used only if ws too small).
__device__ inline v16h load_a_f32g(const float* __restrict__ W, int ldk, int row0,
                                   int kb, int lane) {
  const int row  = row0 + (lane & 15);
  const int koff = (lane >> 4) << 3;
  const float* p = W + row * ldk + kb + koff;
  v16h a;
#pragma unroll
  for (int t = 0; t < 8; ++t) {
    a[t]     = (_Float16)p[t];
    a[8 + t] = (_Float16)p[16 + t];
  }
  return a;
}

// B fragment (32x16) from LDS f16, each column's K contiguous, 16B aligned.
__device__ inline v16h load_b_lds(const _Float16* S, int stride, int col0, int kb,
                                  int lane) {
  const _Float16* p = S + (col0 + (lane & 15)) * stride + kb + ((lane >> 4) << 3);
  const h8 lo = *(const h8*)p;
  const h8 hi = *(const h8*)(p + 16);
  v16h b;
#pragma unroll
  for (int t = 0; t < 8; ++t) { b[t] = lo[t]; b[8 + t] = hi[t]; }
  return b;
}
}  // namespace

__global__ __launch_bounds__(256)
void prep_weights(const float* __restrict__ Wqkv, const float* __restrict__ Wo,
                  const float* __restrict__ Wmlp, _Float16* __restrict__ wsf) {
  const int idx = blockIdx.x * 256 + threadIdx.x;
  if (idx >= (kQT + kOT + kMT) * 32) return;
  const float* W;
  int ldk, KT, tile;
  _Float16* dst;
  if (idx < kQT * 32) {
    W = Wqkv; ldk = 64; KT = 2; dst = wsf + kWqkvOff; tile = idx >> 5;
  } else if (idx < (kQT + kOT) * 32) {
    W = Wo; ldk = 256; KT = 8; dst = wsf + kWoOff; tile = (idx - kQT * 32) >> 5;
  } else {
    W = Wmlp; ldk = 64; KT = 2; dst = wsf + kWmlpOff; tile = (idx - (kQT + kOT) * 32) >> 5;
  }
  const int lane = idx & 31;
  const int mt = tile / KT, kt = tile % KT;
  const int row = mt * 16 + (lane & 15);
  const int kb  = kt * 32 + ((lane >> 4) << 3);
  const float* p = W + row * ldk + kb;
  _Float16* o = dst + (tile * 32 + lane) * 16;
#pragma unroll
  for (int t = 0; t < 8; ++t) {
    o[t]     = (_Float16)p[t];
    o[8 + t] = (_Float16)p[16 + t];
  }
}

template <bool PRE>
__global__ __launch_bounds__(256)
void range_attn_fused(const float* __restrict__ feat, const float* __restrict__ dist,
                      const float* __restrict__ gamma, const float* __restrict__ beta,
                      const float* __restrict__ Wqkv, const float* __restrict__ bqkv,
                      const float* __restrict__ Wo, const float* __restrict__ bo,
                      const float* __restrict__ Wmlp, const float* __restrict__ bmlp,
                      const _Float16* __restrict__ wsf, float* __restrict__ out) {
  __shared__ _Float16 Xh[kCols * kXS];      // bn(x) f16: 11520 B
  __shared__ _Float16 QKVh[kCols * kQS];    // per-head q|k|v f16: 31040 B
  __shared__ _Float16 Oh[kCols * kOS];      // attn-out / mlp input f16: 11520 B
  __shared__ float    Aw[kCols * kCav];     // softmax weights: 1600 B
  __shared__ float    distS[kCav];
  float* Pf = (float*)QKVh;                 // proj f32 aliases QKV buffer

  const int tid  = threadIdx.x;
  const int wave = tid >> 5;
  const int lane = tid & 31;
  const int p0   = blockIdx.x * kPT;
  const float rinv = rsqrtf(1.0f + kEps);

  if (tid < kCav) distS[tid] = dist[tid];

  // Stage bn(feature) tile as f16, column (a*16+p) holds 64 contiguous channels.
  for (int idx = tid; idx < kCols * kC; idx += 256) {
    const int p  = idx & 15;
    const int ac = idx >> 4;
    const int c  = ac & 63;
    const int a  = ac >> 6;
    const float x = feat[(a * kC + c) * kHW + p0 + p];
    Xh[(a * kPT + p) * kXS + c] = (_Float16)(x * (gamma[c] * rinv) + beta[c]);
  }
  __syncthreads();

  v8f pacc[3] = {};  // persistent per-wave accumulators for the output projection

  for (int h = 0; h < kHeads; ++h) {
    // ---- GEMM1: qkv_h(192x80) = Wqkv_h(192x64) @ bnX(64x80).
    for (int j = wave; j < 60; j += 8) {
      const int m = j % 12, n = j / 12;
      const int sec = m >> 2, mq = m & 3;          // 0=Q,1=K,2=V
      const int mt  = sec * 16 + h * 4 + mq;       // row tile in Wqkv
      v8f acc = {};
#pragma unroll
      for (int ks = 0; ks < 2; ++ks) {
        v16h a = PRE ? load_a_ws(wsf + kWqkvOff, 2, mt, ks, lane)
                     : load_a_f32g(Wqkv, kC, mt * 16, ks * 32, lane);
        v16h b = load_b_lds(Xh, kXS, n * 16, ks * 32, lane);
        acc = wmma16x16x32(a, b, acc);
      }
      const int colb = n * 16 + (lane & 15);
      const int hi   = (lane >> 4) << 3;
#pragma unroll
      for (int r = 0; r < 8; ++r) {
        const int lr = mq * 16 + hi + r;           // channel 0..63 within section
        QKVh[colb * kQS + sec * 64 + lr] =
            (_Float16)(acc[r] + bqkv[sec * 256 + h * 64 + lr]);
      }
    }
    __syncthreads();

    // ---- scores(5x5/pixel) + distance bias + softmax; 2 lanes per (p,i) dot.
    if (tid < 2 * kCols) {
      const int u = tid >> 1, half = tid & 1;
      const int p = u / kCav, i = u % kCav;
      const int c0 = half * 32;
      const _Float16* q = &QKVh[(i * kPT + p) * kQS + c0];
      float sc[kCav];
      float qsum = 0.f;
      for (int j = 0; j < kCav; ++j) {
        const _Float16* kk = &QKVh[(j * kPT + p) * kQS + 64 + c0];
        float d = 0.f;
#pragma unroll
        for (int c = 0; c < 32; ++c) {
          const float qc = (float)q[c];
          d += qc * (float)kk[c];
          if (j == 0) qsum += qc;
        }
        sc[j] = d;
      }
      qsum += __shfl_xor(qsum, 1, 32);
#pragma unroll
      for (int j = 0; j < kCav; ++j) sc[j] += __shfl_xor(sc[j], 1, 32);
      const float isq5 = 0.44721359549995794f;  // 1/sqrt(cav)
      float mx = -1e30f;
#pragma unroll
      for (int j = 0; j < kCav; ++j) {
        sc[j] = sc[j] * isq5 + qsum * distS[j];
        mx = fmaxf(mx, sc[j]);
      }
      float se = 0.f;
#pragma unroll
      for (int j = 0; j < kCav; ++j) { sc[j] = __expf(sc[j] - mx); se += sc[j]; }
      const float is = 1.f / se;
      if (half == 0) {
#pragma unroll
        for (int j = 0; j < kCav; ++j) Aw[(p * kCav + i) * kCav + j] = sc[j] * is;
      }
    }
    __syncthreads();

    // ---- out_h = attn @ v -> Oh (f16). 320 jobs: (pixel, agent-i, 16-ch chunk).
    for (int jj = tid; jj < kPT * kCav * 4; jj += 256) {
      const int p    = jj & 15;
      const int rest = jj >> 4;
      const int i    = rest % kCav;
      const int cc   = (rest / kCav) * 16;
      const float* aw = &Aw[(p * kCav + i) * kCav];
      const float a0 = aw[0], a1 = aw[1], a2 = aw[2], a3 = aw[3], a4 = aw[4];
      const _Float16* v0 = &QKVh[(0 * kPT + p) * kQS + 128 + cc];
      const _Float16* v1 = &QKVh[(1 * kPT + p) * kQS + 128 + cc];
      const _Float16* v2 = &QKVh[(2 * kPT + p) * kQS + 128 + cc];
      const _Float16* v3 = &QKVh[(3 * kPT + p) * kQS + 128 + cc];
      const _Float16* v4 = &QKVh[(4 * kPT + p) * kQS + 128 + cc];
      _Float16* o = &Oh[(i * kPT + p) * kOS + cc];
#pragma unroll
      for (int t = 0; t < 16; ++t) {
        o[t] = (_Float16)(a0 * (float)v0[t] + a1 * (float)v1[t] + a2 * (float)v2[t] +
                          a3 * (float)v3[t] + a4 * (float)v4[t]);
      }
    }
    __syncthreads();

    // ---- GEMM2 (accumulated over heads): proj += Wo[:, h*64:+64] @ out_h.
    {
      int slot = 0;
      for (int j = wave; j < 20; j += 8, ++slot) {
        const int m2 = j % 4, n = j / 4;
        v8f acc = pacc[slot];
#pragma unroll
        for (int ks = 0; ks < 2; ++ks) {
          v16h a = PRE ? load_a_ws(wsf + kWoOff, 8, m2, h * 2 + ks, lane)
                       : load_a_f32g(Wo, kHeads * kC, m2 * 16, h * 64 + ks * 32, lane);
          v16h b = load_b_lds(Oh, kOS, n * 16, ks * 32, lane);
          acc = wmma16x16x32(a, b, acc);
        }
        pacc[slot] = acc;
      }
    }
    // No barrier needed here (see barrier chain through the next head's phases).
  }

  // ---- proj (+ b_o) -> LDS f32 (aliases QKV buffer).
  {
    int slot = 0;
    for (int j = wave; j < 20; j += 8, ++slot) {
      const int m2 = j % 4, n = j / 4;
      const int colb = n * 16 + (lane & 15);
      const int hi   = (lane >> 4) << 3;
#pragma unroll
      for (int r = 0; r < 8; ++r) {
        const int c = m2 * 16 + hi + r;
        Pf[colb * kPS + c] = pacc[slot][r] + bo[c];
      }
    }
  }
  __syncthreads();

  // ---- t = bn(bn_x + 2*proj) -> f16 (reuses Oh buffer); bn_x recomputed (L2-hot).
  for (int idx = tid; idx < kCols * kC; idx += 256) {
    const int p  = idx & 15;
    const int ac = idx >> 4;
    const int c  = ac & 63;
    const int a  = ac >> 6;
    const int col = a * kPT + p;
    const float sbn = gamma[c] * rinv;
    const float bnx = feat[(a * kC + c) * kHW + p0 + p] * sbn + beta[c];
    const float tv  = (bnx + 2.f * Pf[col * kPS + c]) * sbn + beta[c];
    Oh[col * kOS + c] = (_Float16)tv;
  }
  __syncthreads();

  // ---- GEMM3 (MLP) + fused final residual: out = bn_x + proj + Wmlp@t + b_mlp.
  for (int j = wave; j < 20; j += 8) {
    const int m2 = j % 4, n = j / 4;  // n is the agent (columns are agent-major)
    v8f acc = {};
#pragma unroll
    for (int ks = 0; ks < 2; ++ks) {
      v16h a = PRE ? load_a_ws(wsf + kWmlpOff, 2, m2, ks, lane)
                   : load_a_f32g(Wmlp, kC, m2 * 16, ks * 32, lane);
      v16h b = load_b_lds(Oh, kOS, n * 16, ks * 32, lane);
      acc = wmma16x16x32(a, b, acc);
    }
    const int pcol = lane & 15;
    const int colb = n * 16 + pcol;
    const int hi   = (lane >> 4) << 3;
#pragma unroll
    for (int r = 0; r < 8; ++r) {
      const int c = m2 * 16 + hi + r;
      const float bnx = feat[(n * kC + c) * kHW + p0 + pcol] * (gamma[c] * rinv) + beta[c];
      out[(n * kC + c) * kHW + p0 + pcol] =
          bnx + Pf[colb * kPS + c] + acc[r] + bmlp[c];
    }
  }
}

extern "C" void kernel_launch(void* const* d_in, const int* in_sizes, int n_in,
                              void* d_out, int out_size, void* d_ws, size_t ws_size,
                              hipStream_t stream) {
  (void)in_sizes; (void)n_in; (void)out_size;
  const float* feat  = (const float*)d_in[0];
  const float* dist  = (const float*)d_in[1];
  const float* gamma = (const float*)d_in[2];
  const float* beta  = (const float*)d_in[3];
  const float* Wqkv  = (const float*)d_in[4];
  const float* bqkv  = (const float*)d_in[5];
  const float* Wo    = (const float*)d_in[6];
  const float* bo    = (const float*)d_in[7];
  const float* Wmlp  = (const float*)d_in[8];
  const float* bmlp  = (const float*)d_in[9];
  float* outp = (float*)d_out;
  _Float16* wsf = (_Float16*)d_ws;

  const bool pre = (d_ws != nullptr) && (ws_size >= (size_t)kWsBytes);
  if (pre) {
    prep_weights<<<dim3(((kQT + kOT + kMT) * 32 + 255) / 256), dim3(256), 0, stream>>>(
        Wqkv, Wo, Wmlp, wsf);
    range_attn_fused<true><<<dim3(kHW / kPT), dim3(256), 0, stream>>>(
        feat, dist, gamma, beta, Wqkv, bqkv, Wo, bo, Wmlp, bmlp, wsf, outp);
  } else {
    range_attn_fused<false><<<dim3(kHW / kPT), dim3(256), 0, stream>>>(
        feat, dist, gamma, beta, Wqkv, bqkv, Wo, bo, Wmlp, bmlp, nullptr, outp);
  }
}
